// PairFeatureInit_80109730005200
// MI455X (gfx1250) — matrix operand order repaired and checked by reading, not compiled, not run on hardware
//
#include <hip/hip_runtime.h>

typedef __bf16 bf16_t;
typedef __attribute__((ext_vector_type(16))) __bf16 bf16x16;
typedef __attribute__((ext_vector_type(8)))  __bf16 bf16x8;
typedef __attribute__((ext_vector_type(8)))  float  f32x8;
typedef int v4i __attribute__((ext_vector_type(4)));

typedef __attribute__((address_space(1))) v4i* as1_v4i_ptr;   // global
typedef __attribute__((address_space(3))) v4i* as3_v4i_ptr;   // LDS

#define B_  2
#define L_  512
#define C_  256
#define P_  128
#define NCP (C_ * P_)   // 32768

union Frag16 { bf16x16 v; bf16x8 h[2]; };

static __device__ inline bf16_t f2bf(float f) {
  unsigned u = __builtin_bit_cast(unsigned, f);
  u += 0x7FFFu + ((u >> 16) & 1u);           // round-to-nearest-even
  unsigned short s = (unsigned short)(u >> 16);
  return __builtin_bit_cast(bf16_t, s);
}

// 16-byte global -> LDS copy. Prefer the gfx1250 async-to-LDS path (ASYNCcnt);
// fall back to a normal load+ds_store pair if the builtin is unavailable.
static __device__ inline void copy_b128_to_lds(const bf16_t* g, bf16_t* l) {
#if __has_builtin(__builtin_amdgcn_global_load_async_to_lds_b128)
  as1_v4i_ptr gp = (as1_v4i_ptr)(unsigned long long)(uintptr_t)g;
  as3_v4i_ptr lp = (as3_v4i_ptr)(unsigned)(uintptr_t)l;
  __builtin_amdgcn_global_load_async_to_lds_b128(gp, lp, 0, 0);
#else
  *(bf16x8*)l = *(const bf16x8*)g;
#endif
}

static __device__ inline void wait_lds_copies() {
#if __has_builtin(__builtin_amdgcn_global_load_async_to_lds_b128)
 #if __has_builtin(__builtin_amdgcn_s_wait_asynccnt)
  __builtin_amdgcn_s_wait_asynccnt(0);
 #else
  asm volatile("s_wait_asynccnt 0x0" ::: "memory");
 #endif
#endif
  __syncthreads();
}

// ---------------- prep: fp32 -> bf16 conversions ----------------

__global__ __launch_bounds__(256) void prep_s_kernel(const float* __restrict__ src,
                                                     bf16_t* __restrict__ dst, int n) {
  int idx = blockIdx.x * 256 + threadIdx.x;
  if (idx < n) dst[idx] = f2bf(src[idx]);
}

// Wt[d][c*P + p] = w_outer[(c*C + d)*P + p]
__global__ __launch_bounds__(256) void prep_w_kernel(const float* __restrict__ w_outer,
                                                     bf16_t* __restrict__ Wt) {
  int idx = blockIdx.x * 256 + threadIdx.x;   // over C_*NCP = 8,388,608
  int d   = idx >> 15;                        // / 32768
  int rem = idx & 32767;                      // c*128 + p
  int c   = rem >> 7;
  int p   = rem & 127;
  Wt[idx] = f2bf(w_outer[(size_t)c * NCP + d * P_ + p]);
}

// ---------------- stage 1: tmp[b][j][c*P+p] = sum_d s[b,j,d] * Wt[d][c*P+p] ----------------
// GEMM: M=512 (j), N=32768, K=256, per batch. 128x128 macro-tile, 8 waves (2x4).

__global__ __launch_bounds__(256) void stage1_gemm(const bf16_t* __restrict__ Sbf,
                                                   const bf16_t* __restrict__ Wt,
                                                   bf16_t* __restrict__ Tmp) {
  __shared__ __attribute__((aligned(16))) bf16_t Alds[128 * C_];  // 64 KB (A tile / out-stage)
  __shared__ __attribute__((aligned(32))) bf16_t Blds[128 * 32];  // 8 KB, col-major [n][k]
  const int nb = blockIdx.x * 128;
  const int mb = blockIdx.y * 128;
  const int b  = blockIdx.z;

  const bf16_t* Ag = Sbf + (size_t)b * (L_ * C_) + (size_t)mb * C_;  // contiguous 64KB tile
  const bf16_t* Bg = Wt;                                             // ldb = NCP
  bf16_t*       D  = Tmp + (size_t)b * (L_ * NCP);                   // ldd = NCP

  const int tid  = threadIdx.x;
  const int lane = tid & 31, wave = tid >> 5;
  const int wm = wave >> 2, wn = wave & 3;
  const int lm = lane & 15, lg = lane >> 4;

  // async-stage the whole A tile (reused by all 8 K-steps)
  #pragma unroll
  for (int it = 0; it < 16; ++it) {
    int off = (tid + it * 256) * 8;
    copy_b128_to_lds(Ag + off, Alds + off);
  }
  wait_lds_copies();

  f32x8 acc[4][2] = {};

  for (int k0 = 0; k0 < C_; k0 += 32) {
    // cooperative load of B tile [32 x 128] -> LDS transposed (col-major)
    #pragma unroll
    for (int it = 0; it < 2; ++it) {
      int q  = tid + it * 256;
      int r  = q >> 4;                  // K row 0..31
      int c0 = (q & 15) * 8;            // N col start
      bf16x8 v = *(const bf16x8*)(Bg + (size_t)(k0 + r) * NCP + nb + c0);
      #pragma unroll
      for (int e = 0; e < 8; ++e) Blds[(c0 + e) * 32 + r] = v[e];
    }
    if (k0 + 32 < C_)
      __builtin_prefetch(Bg + (size_t)(k0 + 32 + (tid >> 4)) * NCP + nb + (tid & 15) * 8, 0, 1);
    __syncthreads();

    Frag16 bfrag[2];
    #pragma unroll
    for (int ni = 0; ni < 2; ++ni) {
      int n = wn * 32 + ni * 16 + lm;
      bfrag[ni].v = *(const bf16x16*)(&Blds[n * 32 + lg * 16]);
    }
    #pragma unroll
    for (int mi = 0; mi < 4; ++mi) {
      int row = wm * 64 + mi * 16 + lm;
      Frag16 a;
      a.h[0] = *(const bf16x8*)(&Alds[row * C_ + k0 + lg * 16]);
      a.h[1] = *(const bf16x8*)(&Alds[row * C_ + k0 + lg * 16 + 8]);
      #pragma unroll
      for (int ni = 0; ni < 2; ++ni)
        acc[mi][ni] = __builtin_amdgcn_wmma_f32_16x16x32_bf16(
            false, a.v, false, bfrag[ni].v, (short)0, acc[mi][ni], false, false);
    }
    __syncthreads();
  }

  // stage output tile in LDS (reuse A region; all A reads done after last barrier),
  // then write fully-coalesced b128 stores into the 67MB tmp stream.
  bf16_t* Dlds = Alds;   // 128x128 bf16 = 32 KB
  #pragma unroll
  for (int mi = 0; mi < 4; ++mi)
    #pragma unroll
    for (int ni = 0; ni < 2; ++ni) {
      int ncol = wn * 32 + ni * 16 + lm;
      #pragma unroll
      for (int r = 0; r < 8; ++r) {
        int mrow = wm * 64 + mi * 16 + r + 8 * lg;   // C/D layout: M = r + 8*lanegrp
        Dlds[mrow * 128 + ncol] = f2bf(acc[mi][ni][r]);
      }
    }
  __syncthreads();
  {
    const int row = tid >> 1, col = (tid & 1) * 64;
    bf16_t* dp = D + (size_t)(mb + row) * NCP + nb + col;
    #pragma unroll
    for (int e = 0; e < 8; ++e)
      *(bf16x8*)(dp + e * 8) = *(const bf16x8*)(&Dlds[row * 128 + col + e * 8]);
  }
}

// ---------------- stage 2: z[b,i,j,p] = sum_c s[b,i,c]*tmp[b,j,c*P+p] + epilogue ----------------
// Per (b, j): GEMM M=512 (i), N=128 (p), K=256 (c). One block = 128 rows of i, all of p.

__global__ __launch_bounds__(256) void stage2_gemm(const bf16_t* __restrict__ Sbf,
                                                   const bf16_t* __restrict__ Tmp,
                                                   const float* __restrict__ pair_bias, // [B][L][L][2]
                                                   const float* __restrict__ mask,      // [B][L]
                                                   const float* __restrict__ b_outer,   // [P]
                                                   const float* __restrict__ pos_embed, // [32][P]
                                                   const float* __restrict__ w_bias,    // [2][P]
                                                   const float* __restrict__ b_bias,    // [P]
                                                   float* __restrict__ Out) {           // [B][L][L][P]
  __shared__ __attribute__((aligned(16))) bf16_t Alds[128 * C_];  // 64 KB
  __shared__ __attribute__((aligned(32))) bf16_t Blds[128 * 32];  // 8 KB
  const int j  = blockIdx.x;
  const int mb = blockIdx.y * 128;
  const int b  = blockIdx.z;

  const bf16_t* Ag = Sbf + (size_t)b * (L_ * C_) + (size_t)mb * C_;
  const bf16_t* Bg = Tmp + ((size_t)b * L_ + j) * NCP;   // [C][P] row-major, ldb = P_

  const int tid  = threadIdx.x;
  const int lane = tid & 31, wave = tid >> 5;
  const int wm = wave >> 2, wn = wave & 3;
  const int lm = lane & 15, lg = lane >> 4;

  #pragma unroll
  for (int it = 0; it < 16; ++it) {
    int off = (tid + it * 256) * 8;
    copy_b128_to_lds(Ag + off, Alds + off);
  }
  wait_lds_copies();

  f32x8 acc[4][2] = {};

  for (int k0 = 0; k0 < C_; k0 += 32) {
    #pragma unroll
    for (int it = 0; it < 2; ++it) {
      int q  = tid + it * 256;
      int r  = q >> 4;
      int c0 = (q & 15) * 8;
      bf16x8 v = *(const bf16x8*)(Bg + (size_t)(k0 + r) * P_ + c0);
      #pragma unroll
      for (int e = 0; e < 8; ++e) Blds[(c0 + e) * 32 + r] = v[e];
    }
    if (k0 + 32 < C_)
      __builtin_prefetch(Bg + (size_t)(k0 + 32 + (tid >> 4)) * P_ + (tid & 15) * 8, 0, 1);
    __syncthreads();

    Frag16 bfrag[2];
    #pragma unroll
    for (int ni = 0; ni < 2; ++ni) {
      int n = wn * 32 + ni * 16 + lm;
      bfrag[ni].v = *(const bf16x16*)(&Blds[n * 32 + lg * 16]);
    }
    #pragma unroll
    for (int mi = 0; mi < 4; ++mi) {
      int row = wm * 64 + mi * 16 + lm;
      Frag16 a;
      a.h[0] = *(const bf16x8*)(&Alds[row * C_ + k0 + lg * 16]);
      a.h[1] = *(const bf16x8*)(&Alds[row * C_ + k0 + lg * 16 + 8]);
      #pragma unroll
      for (int ni = 0; ni < 2; ++ni)
        acc[mi][ni] = __builtin_amdgcn_wmma_f32_16x16x32_bf16(
            false, a.v, false, bfrag[ni].v, (short)0, acc[mi][ni], false, false);
    }
    __syncthreads();
  }

  // fused epilogue + store
  const float mj = mask[b * L_ + j];
  #pragma unroll
  for (int mi = 0; mi < 4; ++mi)
    #pragma unroll
    for (int ni = 0; ni < 2; ++ni) {
      const int p    = wn * 32 + ni * 16 + lm;
      const float ap = b_outer[p] + b_bias[p];
      const float w0 = w_bias[p];
      const float w1 = w_bias[P_ + p];
      #pragma unroll
      for (int r = 0; r < 8; ++r) {
        const int i  = mb + wm * 64 + mi * 16 + r + 8 * lg;
        int rel = i - j;
        rel = (rel > 15 ? 15 : (rel < -15 ? -15 : rel)) + 15;
        const float* pb = pair_bias + (((size_t)b * L_ + i) * L_ + j) * 2;
        float v = acc[mi][ni][r] + ap + pb[0] * w0 + pb[1] * w1 + pos_embed[rel * P_ + p];
        v *= mask[b * L_ + i] * mj;
        Out[(((size_t)b * L_ + i) * L_ + j) * P_ + p] = v;
      }
    }
}

// ---------------- launcher ----------------

extern "C" void kernel_launch(void* const* d_in, const int* in_sizes, int n_in,
                              void* d_out, int out_size, void* d_ws, size_t ws_size,
                              hipStream_t stream) {
  (void)in_sizes; (void)n_in; (void)out_size; (void)ws_size;
  const float* single_repr = (const float*)d_in[0];
  const float* pair_bias   = (const float*)d_in[1];
  const float* mask        = (const float*)d_in[2];
  const float* w_outer     = (const float*)d_in[3];
  const float* b_outer     = (const float*)d_in[4];
  const float* pos_embed   = (const float*)d_in[5];
  const float* w_bias      = (const float*)d_in[6];
  const float* b_bias      = (const float*)d_in[7];
  float* out = (float*)d_out;

  char* ws = (char*)d_ws;
  bf16_t* s_bf = (bf16_t*)ws;                                     // 2*512*256*2   = 0.5 MB
  bf16_t* Wt   = (bf16_t*)(ws + (size_t)524288);                  // 256*32768*2   = 16 MB
  bf16_t* Tmp  = (bf16_t*)(ws + (size_t)524288 + 16777216);       // 2*512*32768*2 = 64 MB

  const int nS = B_ * L_ * C_;                                    // 262144
  prep_s_kernel<<<nS / 256, 256, 0, stream>>>(single_repr, s_bf, nS);
  prep_w_kernel<<<(C_ * NCP) / 256, 256, 0, stream>>>(w_outer, Wt);

  stage1_gemm<<<dim3(NCP / 128, L_ / 128, B_), 256, 0, stream>>>(s_bf, Wt, Tmp);
  stage2_gemm<<<dim3(L_, L_ / 128, B_), 256, 0, stream>>>(
      s_bf, Tmp, pair_bias, mask, b_outer, pos_embed, w_bias, b_bias, out);
}
